// RNN_15693810499788
// MI455X (gfx1250) — matrix operand order repaired and checked by reading, not compiled, not run on hardware
//
#include <hip/hip_runtime.h>
#include <hip/hip_bf16.h>

// ---------------------------------------------------------------------------
// LSTM char-RNN for MI455X (gfx1250), wave32 + WMMA bf16.
// batch=1024, seq=512, UNITS=256, vocab=128.  64 persistent workgroups,
// each owns a 16-row batch tile (h,c private -> no inter-WG sync).
// Recurrent GEMM per step: [16x256] bf16 (LDS) x [256x1024] bf16 (streamed from
// L2 every step, pre-packed into WMMA fragment layout in d_ws) -> fp32 z in LDS.
// An opaque per-step SGPR offset (not an opaque pointer!) blocks LICM from
// hoisting the loop-invariant B fragments while keeping addrspace(1) inference,
// so the loads lower to global_load_b128 saddr-form, not flat_load.
// ---------------------------------------------------------------------------

#define SEQ     512
#define UNITS   256
#define NCHARS  128
#define GATES   1024          // 4*UNITS
#define MT      16            // batch rows per workgroup
#define NWG     64            // 1024 / 16
#define THREADS 512           // 16 waves
#define NTILES  64            // GATES/16
#define KTILES  8             // UNITS/32
#define NT_PER_WAVE 4         // NTILES / 16 waves

typedef __attribute__((ext_vector_type(16))) __bf16 v16bf;
typedef __attribute__((ext_vector_type(8)))  __bf16 v8bf;
typedef __attribute__((ext_vector_type(8)))  float  v8f;

__device__ __forceinline__ unsigned short f32_to_bf16_rne(float f) {
  unsigned u = __float_as_uint(f);
  u += 0x7FFFu + ((u >> 16) & 1u);            // round-to-nearest-even
  return (unsigned short)(u >> 16);
}

__device__ __forceinline__ float fast_sigmoid(float x) {
  return 1.0f / (1.0f + __expf(-x));
}
__device__ __forceinline__ float fast_tanh(float x) {
  return 2.0f / (1.0f + __expf(-2.0f * x)) - 1.0f;
}

// ---------------------------------------------------------------------------
// One-time pre-pack: rec_kernel fp32 [256][1024] -> bf16 WMMA B-fragments.
// Fragment f = nt*8 + kt.  B layout used here (k contiguous per lane half, same order as A):
//   lane L: n = nt*16 + (L&15); elements j=0..15 cover K = kt*32 + (L>>4)*16 + j.
// Stored as 8 dwords/lane, contiguous per fragment -> coalesced b128 loads.
// ---------------------------------------------------------------------------
__global__ __launch_bounds__(32)
void pack_rec(const float* __restrict__ R, unsigned int* __restrict__ ws) {
  const int f    = blockIdx.x;          // 0..511
  const int nt   = f >> 3;
  const int kt   = f & 7;
  const int lane = threadIdx.x;         // 0..31
  const int n     = nt * 16 + (lane & 15);
  const int kbase = kt * 32 + (lane >> 4) * 16;
  unsigned int d[8];
#pragma unroll
  for (int p = 0; p < 8; ++p) {
    unsigned short lo = f32_to_bf16_rne(R[(size_t)(kbase + 2 * p    ) * GATES + n]);
    unsigned short hi = f32_to_bf16_rne(R[(size_t)(kbase + 2 * p + 1) * GATES + n]);
    d[p] = (unsigned)lo | ((unsigned)hi << 16);
  }
  unsigned int* dst = ws + ((size_t)f * 32 + lane) * 8;
#pragma unroll
  for (int p = 0; p < 8; ++p) dst[p] = d[p];
}

// ---------------------------------------------------------------------------
// Persistent LSTM kernel: 64 blocks x 512 threads.
// ---------------------------------------------------------------------------
__global__ __launch_bounds__(THREADS)
void lstm_persistent(const int*   __restrict__ x,      // [1024][512]
                     const float* __restrict__ Wemb,   // [128][1024]
                     const float* __restrict__ bias,   // [1024]
                     const float* __restrict__ dW,     // [256][128]
                     const float* __restrict__ dB,     // [128]
                     const unsigned int* __restrict__ wsFrag,
                     float*       __restrict__ out) {  // [1024][128]
  __shared__ float          zS[MT][GATES];     // 64 KB  gate pre-activations
  __shared__ float          cS[MT][UNITS];     // 16 KB  cell state (fp32 exact)
  __shared__ float          hS[MT][UNITS];     // 16 KB  hidden fp32 (final dense)
  __shared__ unsigned short hB[MT][UNITS];     //  8 KB  hidden bf16 (WMMA A)

  const int tid  = threadIdx.x;
  const int lane = tid & 31;
  const int wave = tid >> 5;
  const int m    = lane & 15;     // A: M index / C: N index
  const int hi   = lane >> 4;     // lane half
  const int row0 = blockIdx.x * MT;

  // ---- init h = c = 0 ----
  for (int e = tid; e < MT * UNITS; e += THREADS) {
    ((float*)cS)[e] = 0.0f;
    ((float*)hS)[e] = 0.0f;
    ((unsigned short*)hB)[e] = 0;
  }

  // per-lane base into the fragment buffer for this wave's 4 N-tiles:
  // frag f = nt*8+kt laid out as [f][lane][8 dwords]
  const unsigned int laneBase =
      (unsigned int)(((wave * NT_PER_WAVE) << 3) * 32 + lane) * 8u;

  // elementwise-phase coordinates: each thread owns 8 contiguous units of one row
  const int erow = tid >> 5;            // 0..15
  const int u0   = (tid & 31) * 8;      // 0..248

#pragma unroll 1
  for (int t = 0; t < SEQ; ++t) {
    __syncthreads();   // hB from previous step ready; zS free

    // Opaque zero offset, renewed every iteration: LICM cannot prove the
    // fragment loads loop-invariant (so no 256-VGPR hoist/spill), but the
    // pointer base stays the kernel argument, so address-space inference
    // keeps the loads in addrspace(1) -> global_load_b128 saddr-form.
    unsigned int zoff = 0;
    asm volatile("" : "+s"(zoff));
    const unsigned int* wB = wsFrag + zoff + laneBase;

    // ---------------- WMMA phase: z = h @ rec_kernel ----------------
    // A fragments (bf16 16x32): lane<16 -> K{0..15}, lane>=16 -> K{16..31}
    v16bf aF[KTILES];
#pragma unroll
    for (int kt = 0; kt < KTILES; ++kt) {
      const v8bf a0 = *(const v8bf*)&hB[m][kt * 32 + hi * 16];
      const v8bf a1 = *(const v8bf*)&hB[m][kt * 32 + hi * 16 + 8];
      aF[kt] = __builtin_shufflevector(a0, a1,
                                       0, 1, 2, 3, 4, 5, 6, 7,
                                       8, 9, 10, 11, 12, 13, 14, 15);
    }

#pragma unroll
    for (int q = 0; q < NT_PER_WAVE; ++q) {
      v8f acc = {};
#pragma unroll
      for (int kt = 0; kt < KTILES; ++kt) {
        // fragment (q,kt) at dword offset ((q*8+kt)*32)*8 from this lane's base
        const v16bf bF = *(const v16bf*)(wB + (((q << 3) | kt) << 8));
        acc = __builtin_amdgcn_wmma_f32_16x16x32_bf16(
            /*neg_a=*/false, aF[kt], /*neg_b=*/false, bF,
            /*c_mod=*/(short)0, acc, /*reuse_a=*/false, /*reuse_b=*/false);
      }
      // C/D layout: lane n = L&15 ; VGPR r -> M = r + 8*(L>>4)
      const int n = (wave * NT_PER_WAVE + q) * 16 + m;
#pragma unroll
      for (int r = 0; r < 8; ++r) zS[hi * 8 + r][n] = acc[r];
    }

    __syncthreads();   // z complete

    // ---------------- elementwise gate phase ----------------
    {
      const int cidx = x[(size_t)(row0 + erow) * SEQ + t];
      const float* wr = Wemb + (size_t)cidx * GATES;
#pragma unroll
      for (int j = 0; j < 8; ++j) {
        const int u = u0 + j;
        const float zi = zS[erow][u]             + wr[u]             + bias[u];
        const float zf = zS[erow][UNITS + u]     + wr[UNITS + u]     + bias[UNITS + u];
        const float zg = zS[erow][2 * UNITS + u] + wr[2 * UNITS + u] + bias[2 * UNITS + u];
        const float zo = zS[erow][3 * UNITS + u] + wr[3 * UNITS + u] + bias[3 * UNITS + u];
        const float ig = fast_sigmoid(zi);
        const float fg = fast_sigmoid(zf);
        const float gg = fast_tanh(zg);
        const float og = fast_sigmoid(zo);
        const float c  = fg * cS[erow][u] + ig * gg;
        const float h  = og * fast_tanh(c);
        cS[erow][u] = c;
        hS[erow][u] = h;
        hB[erow][u] = f32_to_bf16_rne(h);
      }
    }
  }

  __syncthreads();

  // ---------------- final dense: logits = h @ dense_w + dense_b ----------------
  for (int e = tid; e < MT * NCHARS; e += THREADS) {
    const int row = e >> 7;
    const int col = e & (NCHARS - 1);
    float s = dB[col];
#pragma unroll 4
    for (int u = 0; u < UNITS; ++u) s += hS[row][u] * dW[(size_t)u * NCHARS + col];
    zS[row][col] = s;   // reuse zS for logits
  }
  __syncthreads();

  // ---------------- softmax per row ----------------
  if (tid < MT) {
    const int row = tid;
    float mx = -1e30f;
    for (int c = 0; c < NCHARS; ++c) mx = fmaxf(mx, zS[row][c]);
    float sum = 0.0f;
    for (int c = 0; c < NCHARS; ++c) {
      const float e2 = __expf(zS[row][c] - mx);
      zS[row][c] = e2;
      sum += e2;
    }
    const float inv = 1.0f / sum;
    for (int c = 0; c < NCHARS; ++c)
      out[(size_t)(row0 + row) * NCHARS + c] = zS[row][c] * inv;
  }
}

// ---------------------------------------------------------------------------
extern "C" void kernel_launch(void* const* d_in, const int* in_sizes, int n_in,
                              void* d_out, int out_size, void* d_ws, size_t ws_size,
                              hipStream_t stream) {
  const int*   x     = (const int*)  d_in[0];   // [1024][512] int32
  const float* Wemb  = (const float*)d_in[1];   // [128][1024]
  const float* R     = (const float*)d_in[2];   // [256][1024]
  const float* bias  = (const float*)d_in[3];   // [1024]
  const float* dW    = (const float*)d_in[4];   // [256][128]
  const float* dB    = (const float*)d_in[5];   // [128]
  float*       out   = (float*)d_out;           // [1024][128]
  unsigned int* ws   = (unsigned int*)d_ws;     // 512 KB bf16 fragment buffer

  // one-time weight pre-pack (fp32 -> bf16 WMMA fragments), then persistent LSTM
  pack_rec<<<NTILES * KTILES, 32, 0, stream>>>(R, ws);
  lstm_persistent<<<NWG, THREADS, 0, stream>>>(x, Wemb, bias, dW, dB, ws, out);
}
